// TorchModel_37288906064223
// MI455X (gfx1250) — matrix operand order repaired and checked by reading, not compiled, not run on hardware
//
#include <hip/hip_runtime.h>
#include <hip/hip_bf16.h>
#include <math.h>

// ---------------------------------------------------------------------------
// MI455X (gfx1250). wave32. GEMMs on v_wmma_f32_16x16x32_f16, A-tile staged to
// LDS via the Tensor Data Mover (tensor_load_to_lds + s_wait_tensorcnt),
// weights pre-transposed to [N,K] f16 so B fragments are direct vector loads.
// ---------------------------------------------------------------------------

typedef __attribute__((ext_vector_type(16))) _Float16 v16h;
typedef __attribute__((ext_vector_type(8)))  _Float16 v8h;
typedef __attribute__((ext_vector_type(8)))  float    v8f;
typedef unsigned int  u32x4 __attribute__((ext_vector_type(4)));
typedef int           i32x4 __attribute__((ext_vector_type(4)));
typedef int           i32x8 __attribute__((ext_vector_type(8)));

#define NODES   100000
#define EDGES   1600000
#define PAIRS   500000
#define KHOPS   6

#if defined(__has_builtin)
#if __has_builtin(__builtin_amdgcn_tensor_load_to_lds)
#define HAVE_TDM 1
#endif
#if __has_builtin(__builtin_amdgcn_s_wait_tensorcnt)
#define HAVE_TENSORCNT 1
#endif
#endif

__device__ __forceinline__ float gelu_erf(float x) {
    return 0.5f * x * (1.0f + erff(x * 0.70710678118654752f));
}

// -------------------------------- utility ----------------------------------

__global__ void fill_f32_kernel(float* __restrict__ p, float v, long long n) {
    long long i = (long long)blockIdx.x * blockDim.x + threadIdx.x;
    if (i < n) p[i] = v;
}

__global__ void f32_to_f16_kernel(const float* __restrict__ in, _Float16* __restrict__ out, long long n) {
    long long i = (long long)blockIdx.x * blockDim.x + threadIdx.x;
    if (i < n) out[i] = (_Float16)in[i];
}

// WT[n*K + k] = (f16) W[k*N + n]   (weights are tiny: do it once)
__global__ void transpose_w_f16_kernel(const float* __restrict__ W, _Float16* __restrict__ WT,
                                       int K, int N) {
    long long i = (long long)blockIdx.x * blockDim.x + threadIdx.x;
    if (i >= (long long)K * N) return;
    int n = (int)(i / K);
    int k = (int)(i - (long long)n * K);
    WT[i] = (_Float16)W[(size_t)k * N + n];
}

// ------------------------- batch-norm helpers ------------------------------

__global__ void bn_node_stats_kernel(const float* __restrict__ X, float* __restrict__ stats) {
    int gid = blockIdx.x * blockDim.x + threadIdx.x;
    int c = gid & 127;
    int rstride = (gridDim.x * blockDim.x) >> 7;
    float s = 0.f, q = 0.f;
    for (int r = gid >> 7; r < NODES; r += rstride) {
        float v = X[(size_t)r * 128 + c];
        s += v; q += v * v;
    }
    atomicAdd(&stats[c], s);
    atomicAdd(&stats[128 + c], q);
}

__global__ void bn_finalize_kernel(const float* __restrict__ stats,
                                   const float* __restrict__ g, const float* __restrict__ b,
                                   float* __restrict__ scale, float* __restrict__ shift,
                                   int C, float count, float eps) {
    int c = blockIdx.x * blockDim.x + threadIdx.x;
    if (c >= C) return;
    float mu  = stats[c] / count;
    float var = stats[C + c] / count - mu * mu;
    float sc  = rsqrtf(var + eps) * g[c];
    scale[c] = sc;
    shift[c] = b[c] - mu * sc;
}

__global__ void bn_node_apply_kernel(const float* __restrict__ X,
                                     const float* __restrict__ scale, const float* __restrict__ shift,
                                     _Float16* __restrict__ xh) {
    long long i = (long long)blockIdx.x * blockDim.x + threadIdx.x;
    if (i >= (long long)NODES * 128) return;
    int c = (int)(i & 127);
    xh[i] = (_Float16)(X[i] * scale[c] + shift[c]);
}

// ----------------------------- WMMA GEMM -----------------------------------
// C[M,N] = act( A[M,K](f16) * B[K,N] + bias ), B given pre-transposed BT[N,K].
// Requirements: N % 64 == 0, K % 32 == 0 (true at every call site).
// Block: 128 threads = 4 waves; tile 128(M) x 64(N); per wave 2x4 wmma tiles.
// A tile (128x32 f16) staged to LDS by the Tensor Data Mover each K-step.

__global__ __launch_bounds__(128) void gemm_wmma_f16_kernel(
    const _Float16* __restrict__ A, int lda,
    const _Float16* __restrict__ BT,   // [N, K] row-major f16
    const float* __restrict__ bias,
    const float* __restrict__ Cin,     // f32 accumulate source (ld = N) or nullptr
    float* __restrict__ outF32,        // optional f32 output (ld = N)
    _Float16* __restrict__ outF16,     // optional f16 output (ld = N)
    int M, int N, int K, int act)      // act: 0=none, 1=gelu
{
    // row stride 40 halves (TDM pads 4 dwords every 16 dwords) -> conflict-free b128
    __shared__ _Float16 As[128][40] __attribute__((aligned(16)));

    const int n0   = blockIdx.x * 64;
    const int m0   = blockIdx.y * 128;
    const int tid  = threadIdx.x;
    const int lane = tid & 31;
    const int wid  = tid >> 5;
    const int hi   = lane >> 4;
    const int l16  = lane & 15;

    v8f acc[2][4];
#pragma unroll
    for (int s = 0; s < 2; ++s)
#pragma unroll
        for (int t = 0; t < 4; ++t)
#pragma unroll
            for (int j = 0; j < 8; ++j) acc[s][t][j] = 0.f;

    if (Cin) {
#pragma unroll
        for (int s = 0; s < 2; ++s)
#pragma unroll
            for (int t = 0; t < 4; ++t) {
                int col = n0 + t * 16 + l16;
#pragma unroll
                for (int j = 0; j < 8; ++j) {
                    int grow = m0 + wid * 32 + s * 16 + hi * 8 + j;
                    if (grow < M) acc[s][t][j] = Cin[(size_t)grow * N + col];
                }
            }
    }

    for (int k0 = 0; k0 < K; k0 += 32) {
#ifdef HAVE_TDM
        if (tid < 32) {
            // ---- Tensor DMA: load A[m0:m0+128, k0:k0+32] (f16) into As ----
            unsigned long long gaddr =
                (unsigned long long)(uintptr_t)(A + (size_t)m0 * lda + k0);
            unsigned ldsOff = (unsigned)(uintptr_t)&As[0][0];  // LDS byte offset
            unsigned rows = (unsigned)((M - m0) < 128 ? (M - m0) : 128);

            u32x4 g0;
            g0[0] = 1u;                                   // count=1, user mode
            g0[1] = ldsOff;                               // lds_addr
            g0[2] = (unsigned)(gaddr & 0xFFFFFFFFull);    // global_addr[31:0]
            g0[3] = (unsigned)((gaddr >> 32) & 0x01FFFFFFull) | (2u << 30); // type=2

            unsigned dim0 = 32u, tile0 = 32u, tile1 = 128u;
            unsigned long long str0 = (unsigned long long)lda;  // data_size units
            i32x8 g1;
            g1[0] = (int)((1u << 16)              // data_size = 2 bytes
                        | (1u << 20)              // pad_enable
                        | (3u << 22)              // pad_interval: 16 dwords
                        | (3u << 25));            // pad_amount: 4 dwords
            g1[1] = (int)((dim0 & 0xFFFFu) << 16);                      // tensor_dim0 lo
            g1[2] = (int)((dim0 >> 16) | ((rows & 0xFFFFu) << 16));     // dim0 hi | dim1 lo
            g1[3] = (int)((rows >> 16) | (tile0 << 16));                // dim1 hi | tile_dim0
            g1[4] = (int)(tile1 & 0xFFFFu);                             // tile_dim1
            g1[5] = (int)(str0 & 0xFFFFFFFFull);                        // dim0_stride lo
            g1[6] = (int)((str0 >> 32) & 0xFFFFull);                    // dim0_stride hi
            g1[7] = 0;
            i32x4 gz4 = {0, 0, 0, 0};
            i32x8 gz8 = {0, 0, 0, 0, 0, 0, 0, 0};
            __builtin_amdgcn_tensor_load_to_lds(g0, g1, gz4, gz4, gz8, 0);
#ifdef HAVE_TENSORCNT
            __builtin_amdgcn_s_wait_tensorcnt(0);
#else
            asm volatile("s_wait_tensorcnt 0x0" ::: "memory");
#endif
            asm volatile("" ::: "memory");
        }
#else
        // ---- fallback: cooperative vectorized global->LDS copy ----
#pragma unroll
        for (int i = 0; i < 4; ++i) {
            int c = tid + i * 128;            // 0..511 chunks of 8 halves
            int r = c >> 2;
            int col = (c & 3) * 8;
            v8h v = {};
            int grow = m0 + r;
            if (grow < M) v = *(const v8h*)(A + (size_t)grow * lda + k0 + col);
            *(v8h*)&As[r][col] = v;
        }
#endif
        __syncthreads();

        // B fragments straight from global (pre-transposed, cache-resident)
        v16h bf[4];
#pragma unroll
        for (int t = 0; t < 4; ++t) {
            const _Float16* bp = BT + (size_t)(n0 + t * 16 + l16) * K + k0 + hi * 8;
            v8h blo = *(const v8h*)bp;
            v8h bhi = *(const v8h*)(bp + 16);
#pragma unroll
            for (int e = 0; e < 8; ++e) { bf[t][e] = blo[e]; bf[t][e + 8] = bhi[e]; }
        }

#pragma unroll
        for (int s = 0; s < 2; ++s) {
            int arow = wid * 32 + s * 16 + l16;
            v8h alo = *(const v8h*)&As[arow][hi * 8];
            v8h ahi2 = *(const v8h*)&As[arow][16 + hi * 8];
            v16h af;
#pragma unroll
            for (int e = 0; e < 8; ++e) { af[e] = alo[e]; af[e + 8] = ahi2[e]; }
#pragma unroll
            for (int t = 0; t < 4; ++t) {
                acc[s][t] = __builtin_amdgcn_wmma_f32_16x16x32_f16(
                    false, af, false, bf[t], (short)0, acc[s][t], false, false);
            }
        }
        __syncthreads();
    }

    // epilogue
#pragma unroll
    for (int s = 0; s < 2; ++s)
#pragma unroll
        for (int t = 0; t < 4; ++t) {
            int col = n0 + t * 16 + l16;
            float bv = bias ? bias[col] : 0.f;
#pragma unroll
            for (int j = 0; j < 8; ++j) {
                int grow = m0 + wid * 32 + s * 16 + hi * 8 + j;
                if (grow < M) {
                    float v = acc[s][t][j] + bv;
                    if (act == 1) v = gelu_erf(v);
                    size_t o = (size_t)grow * N + col;
                    if (outF32) outF32[o] = v;
                    if (outF16) outF16[o] = (_Float16)v;
                }
            }
        }
}

// ------------------------------ TAGConv ------------------------------------

__global__ void degree_kernel(const int* __restrict__ dst, float* __restrict__ deg) {
    int e = blockIdx.x * blockDim.x + threadIdx.x;
    if (e < EDGES) atomicAdd(&deg[dst[e]], 1.0f);
}

__global__ void dinv_kernel(const float* __restrict__ deg, float* __restrict__ dinv) {
    int n = blockIdx.x * blockDim.x + threadIdx.x;
    if (n >= NODES) return;
    float d = deg[n];
    dinv[n] = (d > 0.f) ? rsqrtf(fmaxf(d, 1.0f)) : 0.f;
}

// one wave per edge, float4 per lane: out[dst] += hk[src] * dinv[src]*dinv[dst]
__global__ void spmm_hop_kernel(const float* __restrict__ hk,
                                const int* __restrict__ src, const int* __restrict__ dst,
                                const float* __restrict__ dinv, float* __restrict__ out) {
    long long t = (long long)blockIdx.x * blockDim.x + threadIdx.x;
    long long e = t >> 5;
    if (e >= EDGES) return;
    int lane = (int)(t & 31);
    int s = src[e], d = dst[e];
    float w = dinv[s] * dinv[d];
    const float4 hv = *(const float4*)(hk + (size_t)s * 128 + lane * 4);
    float* o = out + (size_t)d * 128 + lane * 4;
    atomicAdd(o + 0, hv.x * w);
    atomicAdd(o + 1, hv.y * w);
    atomicAdd(o + 2, hv.z * w);
    atomicAdd(o + 3, hv.w * w);
}

// A256[n] = [ x16(n) | f16(gelu(tagOut(n) + tag_bias)) ]
__global__ void build_a256_kernel(const _Float16* __restrict__ x16,
                                  const float* __restrict__ tagOut,
                                  const float* __restrict__ tag_bias,
                                  _Float16* __restrict__ A256) {
    long long i = (long long)blockIdx.x * blockDim.x + threadIdx.x;
    if (i >= (long long)NODES * 128) return;
    long long n = i >> 7;
    int c = (int)(i & 127);
    A256[n * 256 + c] = x16[i];
    A256[n * 256 + 128 + c] = (_Float16)gelu_erf(tagOut[i] + tag_bias[c]);
}

// --------------------------- node classifier end ---------------------------

__global__ void end_softmax_kernel(const _Float16* __restrict__ cl16,
                                   const float* __restrict__ W,   // [128,8]
                                   const float* __restrict__ b,   // [8]
                                   float* __restrict__ out) {
    __shared__ float ws[128 * 8];
    __shared__ float bs[8];
    for (int i = threadIdx.x; i < 128 * 8; i += blockDim.x) ws[i] = W[i];
    if (threadIdx.x < 8) bs[threadIdx.x] = b[threadIdx.x];
    __syncthreads();
    int n = blockIdx.x * blockDim.x + threadIdx.x;
    if (n >= NODES) return;
    float lg[8];
#pragma unroll
    for (int j = 0; j < 8; ++j) lg[j] = bs[j];
    const _Float16* row = cl16 + (size_t)n * 128;
    for (int k = 0; k < 128; ++k) {
        float x = (float)row[k];
#pragma unroll
        for (int j = 0; j < 8; ++j) lg[j] += x * ws[k * 8 + j];
    }
    float m = lg[0];
#pragma unroll
    for (int j = 1; j < 8; ++j) m = fmaxf(m, lg[j]);
    float sum = 0.f;
#pragma unroll
    for (int j = 0; j < 8; ++j) { lg[j] = expf(lg[j] - m); sum += lg[j]; }
    float inv = 1.f / sum;
#pragma unroll
    for (int j = 0; j < 8; ++j) out[(size_t)n * 8 + j] = lg[j] * inv;
}

// ------------------------------ edge stage ---------------------------------

__device__ __forceinline__ float omega_val(long long p, int c,
                                           const float* __restrict__ a,
                                           const float* __restrict__ X,
                                           const float* __restrict__ Y,
                                           const int* __restrict__ i0,
                                           const int* __restrict__ i1) {
    if (c < 128) return a[(size_t)i0[p] * 128 + c];
    if (c < 256) return a[(size_t)i1[p] * 128 + (c - 128)];
    if (c < 384) return X[(size_t)i0[p] * 128 + (c - 256)];
    if (c < 512) return X[(size_t)i1[p] * 128 + (c - 384)];
    return Y[(size_t)p * 32 + (c - 512)];
}

__global__ void edge_stats_kernel(const float* __restrict__ a, const float* __restrict__ X,
                                  const float* __restrict__ Y,
                                  const int* __restrict__ i0, const int* __restrict__ i1,
                                  float* __restrict__ stats) {
    int gid = blockIdx.x * blockDim.x + threadIdx.x;
    int c = gid % 544;
    int pstride = (gridDim.x * blockDim.x) / 544;
    float s = 0.f, q = 0.f;
    for (long long p = gid / 544; p < PAIRS; p += pstride) {
        float v = omega_val(p, c, a, X, Y, i0, i1);
        s += v; q += v * v;
    }
    atomicAdd(&stats[c], s);
    atomicAdd(&stats[544 + c], q);
}

__global__ void omega_build_kernel(const float* __restrict__ a, const float* __restrict__ X,
                                   const float* __restrict__ Y,
                                   const int* __restrict__ i0, const int* __restrict__ i1,
                                   const float* __restrict__ scale, const float* __restrict__ shift,
                                   _Float16* __restrict__ Om) {
    long long i = (long long)blockIdx.x * blockDim.x + threadIdx.x;
    if (i >= (long long)PAIRS * 544) return;
    long long p = i / 544;
    int c = (int)(i - p * 544);
    float v = omega_val(p, c, a, X, Y, i0, i1) * scale[c] + shift[c];
    Om[i] = (_Float16)v;
}

__global__ void edge_final_kernel(const _Float16* __restrict__ O2,
                                  const float* __restrict__ W,   // [256]
                                  const float* __restrict__ b,   // [1]
                                  float* __restrict__ out) {
    __shared__ float ws[256];
    for (int i = threadIdx.x; i < 256; i += blockDim.x) ws[i] = W[i];
    __syncthreads();
    long long p = (long long)blockIdx.x * blockDim.x + threadIdx.x;
    if (p >= PAIRS) return;
    const _Float16* row = O2 + (size_t)p * 256;
    float acc = b[0];
    for (int k = 0; k < 256; ++k) acc += (float)row[k] * ws[k];
    out[p] = 1.0f / (1.0f + expf(-acc));
}

// ------------------------------- driver ------------------------------------

static inline int cdiv(long long a, long long b) { return (int)((a + b - 1) / b); }

extern "C" void kernel_launch(void* const* d_in, const int* in_sizes, int n_in,
                              void* d_out, int out_size, void* d_ws, size_t ws_size,
                              hipStream_t stream) {
    const float* X        = (const float*)d_in[0];
    const float* Y        = (const float*)d_in[1];
    const int*   EI       = (const int*)d_in[2];
    const int*   I0       = (const int*)d_in[3];
    const int*   I1       = (const int*)d_in[4];
    const float* bn_g     = (const float*)d_in[5];
    const float* bn_b     = (const float*)d_in[6];
    const float* Wlin     = (const float*)d_in[7];
    const float* blin     = (const float*)d_in[8];
    const float* tagWs    = (const float*)d_in[9];
    const float* tag_bias = (const float*)d_in[10];
    const float* nl_W0    = (const float*)d_in[11];
    const float* nl_b0    = (const float*)d_in[12];
    const float* nl_W1    = (const float*)d_in[13];
    const float* nl_b1    = (const float*)d_in[14];
    const float* cl_W     = (const float*)d_in[15];
    const float* cl_b     = (const float*)d_in[16];
    const float* end_W    = (const float*)d_in[17];
    const float* end_b    = (const float*)d_in[18];
    const float* beg      = (const float*)d_in[19];
    const float* beb      = (const float*)d_in[20];
    const float* el_W0    = (const float*)d_in[21];
    const float* el_b0    = (const float*)d_in[22];
    const float* el_W1    = (const float*)d_in[23];
    const float* el_b1    = (const float*)d_in[24];
    const float* el_end_W = (const float*)d_in[25];
    const float* el_end_b = (const float*)d_in[26];

    const int* src = EI;
    const int* dst = EI + EDGES;

    char* ws = (char*)d_ws;
    size_t off = 0;
    auto alloc = [&](size_t bytes) -> char* {
        char* p = ws + off;
        off += (bytes + 255) & ~(size_t)255;
        return p;
    };
    float*    nodeStats = (float*)alloc(256 * 4);
    float*    nodeScale = (float*)alloc(128 * 4);
    float*    nodeShift = (float*)alloc(128 * 4);
    float*    deg       = (float*)alloc((size_t)NODES * 4);
    float*    dinv      = (float*)alloc((size_t)NODES * 4);
    float*    edgeStats = (float*)alloc(1088 * 4);
    float*    edgeScale = (float*)alloc(544 * 4);
    float*    edgeShift = (float*)alloc(544 * 4);
    // pre-transposed f16 weights: WT[N,K]
    _Float16* wt_lin    = (_Float16*)alloc(16384 * 2);
    _Float16* wt_tag    = (_Float16*)alloc(114688 * 2);         // 7 x [128,128]
    _Float16* wt_nl0    = (_Float16*)alloc(65536 * 2);
    _Float16* wt_nl1    = (_Float16*)alloc(32768 * 2);
    _Float16* wt_cl     = (_Float16*)alloc(16384 * 2);
    _Float16* wt_el0    = (_Float16*)alloc((size_t)544 * 512 * 2);
    _Float16* wt_el1    = (_Float16*)alloc((size_t)512 * 256 * 2);
    _Float16* x16       = (_Float16*)alloc((size_t)NODES * 128 * 2);
    _Float16* h16       = (_Float16*)alloc((size_t)NODES * 128 * 2);
    float*    tagOut    = (float*)alloc((size_t)NODES * 128 * 4);
    float*    hkA       = (float*)alloc((size_t)NODES * 128 * 4);
    float*    hkB       = (float*)alloc((size_t)NODES * 128 * 4);
    _Float16* hk16      = (_Float16*)alloc((size_t)NODES * 128 * 2);
    _Float16* A256      = (_Float16*)alloc((size_t)NODES * 256 * 2);
    _Float16* T256      = (_Float16*)alloc((size_t)NODES * 256 * 2);
    float*    a_f32     = (float*)alloc((size_t)NODES * 128 * 4);
    _Float16* a16       = (_Float16*)alloc((size_t)NODES * 128 * 2);
    _Float16* cl16      = (_Float16*)alloc((size_t)NODES * 128 * 2);
    _Float16* Om16      = (_Float16*)alloc((size_t)PAIRS * 544 * 2);
    _Float16* O1_16     = (_Float16*)alloc((size_t)PAIRS * 512 * 2);
    _Float16* O2_16     = (_Float16*)alloc((size_t)PAIRS * 256 * 2);
    (void)ws_size;

    float* out_classes = (float*)d_out;                 // [NODES, 8]
    float* out_epred   = (float*)d_out + (size_t)NODES * 8;

    const long long NF = (long long)NODES * 128;

    // ---- node batch norm ----
    fill_f32_kernel<<<1, 256, 0, stream>>>(nodeStats, 0.f, 256);
    bn_node_stats_kernel<<<128, 256, 0, stream>>>(X, nodeStats);
    bn_finalize_kernel<<<1, 128, 0, stream>>>(nodeStats, bn_g, bn_b, nodeScale, nodeShift,
                                              128, (float)NODES, 1e-5f);
    bn_node_apply_kernel<<<cdiv(NF, 256), 256, 0, stream>>>(X, nodeScale, nodeShift, x16);

    // ---- transpose+convert all weights to WT[N,K] f16 ----
    transpose_w_f16_kernel<<<cdiv(16384, 256), 256, 0, stream>>>(Wlin, wt_lin, 128, 128);
    for (int k = 0; k <= KHOPS; ++k)
        transpose_w_f16_kernel<<<cdiv(16384, 256), 256, 0, stream>>>(
            tagWs + (size_t)k * 16384, wt_tag + (size_t)k * 16384, 128, 128);
    transpose_w_f16_kernel<<<cdiv(65536, 256), 256, 0, stream>>>(nl_W0, wt_nl0, 256, 256);
    transpose_w_f16_kernel<<<cdiv(32768, 256), 256, 0, stream>>>(nl_W1, wt_nl1, 256, 128);
    transpose_w_f16_kernel<<<cdiv(16384, 256), 256, 0, stream>>>(cl_W, wt_cl, 128, 128);
    transpose_w_f16_kernel<<<cdiv(544 * 512, 256), 256, 0, stream>>>(el_W0, wt_el0, 544, 512);
    transpose_w_f16_kernel<<<cdiv(512 * 256, 256), 256, 0, stream>>>(el_W1, wt_el1, 512, 256);

    // ---- h = gelu(x @ Wlin + b); keep f32 copy as hop-0 state ----
    {
        dim3 g(2, cdiv(NODES, 128));
        gemm_wmma_f16_kernel<<<g, 128, 0, stream>>>(x16, 128, wt_lin, blin,
                                                    nullptr, hkA, h16, NODES, 128, 128, 1);
    }

    // ---- gcn normalization ----
    fill_f32_kernel<<<cdiv(NODES, 256), 256, 0, stream>>>(deg, 0.f, NODES);
    degree_kernel<<<cdiv(EDGES, 256), 256, 0, stream>>>(dst, deg);
    dinv_kernel<<<cdiv(NODES, 256), 256, 0, stream>>>(deg, dinv);

    // ---- tagOut = h @ Ws[0] ----
    {
        dim3 g(2, cdiv(NODES, 128));
        gemm_wmma_f16_kernel<<<g, 128, 0, stream>>>(h16, 128, wt_tag, nullptr,
                                                    nullptr, tagOut, nullptr, NODES, 128, 128, 0);
    }

    // ---- 6 hops: SpMM scatter + WMMA accumulate ----
    float* hkCur = hkA;
    float* hkNxt = hkB;
    for (int k = 1; k <= KHOPS; ++k) {
        fill_f32_kernel<<<cdiv(NF, 256), 256, 0, stream>>>(hkNxt, 0.f, NF);
        spmm_hop_kernel<<<cdiv((long long)EDGES * 32, 256), 256, 0, stream>>>(hkCur, src, dst, dinv, hkNxt);
        f32_to_f16_kernel<<<cdiv(NF, 256), 256, 0, stream>>>(hkNxt, hk16, NF);
        dim3 g(2, cdiv(NODES, 128));
        gemm_wmma_f16_kernel<<<g, 128, 0, stream>>>(hk16, 128, wt_tag + (size_t)k * 16384,
                                                    nullptr, tagOut, tagOut, nullptr,
                                                    NODES, 128, 128, 0);
        float* t = hkCur; hkCur = hkNxt; hkNxt = t;
    }

    // ---- node MLP head ----
    build_a256_kernel<<<cdiv(NF, 256), 256, 0, stream>>>(x16, tagOut, tag_bias, A256);
    {
        dim3 g0(4, cdiv(NODES, 128));
        gemm_wmma_f16_kernel<<<g0, 128, 0, stream>>>(A256, 256, wt_nl0, nl_b0,
                                                     nullptr, nullptr, T256, NODES, 256, 256, 1);
        dim3 g1(2, cdiv(NODES, 128));
        gemm_wmma_f16_kernel<<<g1, 128, 0, stream>>>(T256, 256, wt_nl1, nl_b1,
                                                     nullptr, a_f32, a16, NODES, 128, 256, 1);
        gemm_wmma_f16_kernel<<<g1, 128, 0, stream>>>(a16, 128, wt_cl, cl_b,
                                                     nullptr, nullptr, cl16, NODES, 128, 128, 1);
    }
    end_softmax_kernel<<<cdiv(NODES, 256), 256, 0, stream>>>(cl16, end_W, end_b, out_classes);

    // ---- edge stage ----
    fill_f32_kernel<<<cdiv(1088, 256), 256, 0, stream>>>(edgeStats, 0.f, 1088);
    edge_stats_kernel<<<544, 256, 0, stream>>>(a_f32, X, Y, I0, I1, edgeStats);
    bn_finalize_kernel<<<cdiv(544, 128), 128, 0, stream>>>(edgeStats, beg, beb,
                                                           edgeScale, edgeShift,
                                                           544, (float)PAIRS, 1.0f);
    omega_build_kernel<<<cdiv((long long)PAIRS * 544, 256), 256, 0, stream>>>(
        a_f32, X, Y, I0, I1, edgeScale, edgeShift, Om16);
    {
        dim3 g0(8, cdiv(PAIRS, 128));
        gemm_wmma_f16_kernel<<<g0, 128, 0, stream>>>(Om16, 544, wt_el0, el_b0,
                                                     nullptr, nullptr, O1_16, PAIRS, 512, 544, 1);
        dim3 g1(4, cdiv(PAIRS, 128));
        gemm_wmma_f16_kernel<<<g1, 128, 0, stream>>>(O1_16, 512, wt_el1, el_b1,
                                                     nullptr, nullptr, O2_16, PAIRS, 256, 512, 1);
    }
    edge_final_kernel<<<cdiv(PAIRS, 256), 256, 0, stream>>>(O2_16, el_end_W, el_end_b, out_epred);

    (void)in_sizes; (void)n_in; (void)out_size;
}
